// CompressiveTransformer_5669356832590
// MI455X (gfx1250) — matrix operand order repaired and checked by reading, not compiled, not run on hardware
//
#include <hip/hip_runtime.h>
#include <hip/hip_bf16.h>

typedef __bf16 bhalf;
typedef __bf16 v16bf __attribute__((ext_vector_type(16)));
typedef __bf16 v8bf  __attribute__((ext_vector_type(8)));
typedef float  v8f   __attribute__((ext_vector_type(8)));

// ---------------------------------------------------------------------------
// f32 -> bf16 conversion, batched-strided:  dst[by*dbs + i] = src[by*sbs + i]
// ---------------------------------------------------------------------------
__global__ void cvt_f32_bf16_kernel(const float* __restrict__ src, bhalf* __restrict__ dst,
                                    long long n, long long sbs, long long dbs) {
  long long i = (long long)blockIdx.x * 256 + threadIdx.x;
  if (i < n)
    dst[(long long)blockIdx.y * dbs + i] = (bhalf)src[(long long)blockIdx.y * sbs + i];
}

// conv_w (O,I,R) -> bf16 (O, R*E + I) so conv becomes GEMM over K = R*E
__global__ void repack_convw_kernel(const float* __restrict__ w, bhalf* __restrict__ o, long long n) {
  long long idx = (long long)blockIdx.x * 256 + threadIdx.x;
  if (idx < n) {
    long long oo = idx >> 12;        // /4096
    int rem = (int)(idx & 4095);
    int r = rem >> 10;               // /1024
    int i = rem & 1023;
    o[idx] = (bhalf)w[(oo << 12) + (long long)i * 4 + r];
  }
}

// ---------------------------------------------------------------------------
// Generic batched WMMA GEMM:  C = alpha * A * B^T (+ bias)
// A: bf16 [M,K] row stride lda; B: bf16 [N,K] row stride ldb; z = b*H + h.
// Block = 256 threads = 8 waves, tile 128x128, wave tile 32x64, K-step 64.
// A/B tiles are streamed global->LDS with GLOBAL_LOAD_ASYNC_TO_LDS_B128
// (ASYNCcnt), double-buffered; WMMAs consume from LDS (ds_load_b128).
// ---------------------------------------------------------------------------
struct GemmP {
  const bhalf* A; const bhalf* B;
  float* Cf; bhalf* Cb;
  const float* bias;
  long long a_bs, a_hs, b_bs, b_hs, c_bs, c_hs;
  int lda, ldb, ldc, K, H;
  float alpha;
};

#define TPAD 72  // 64 + 8 pad: 144B row stride -> conflict-free b128 LDS reads

__global__ __launch_bounds__(256) void wmma_gemm_kernel(GemmP p) {
  __shared__ bhalf At[2][128][TPAD];
  __shared__ bhalf Bt[2][128][TPAD];

  const int z  = blockIdx.z;
  const int bb = z / p.H, hh = z % p.H;
  const bhalf* __restrict__ A  = p.A + (long long)bb * p.a_bs + (long long)hh * p.a_hs
                               + (long long)blockIdx.y * 128 * p.lda;
  const bhalf* __restrict__ Bm = p.B + (long long)bb * p.b_bs + (long long)hh * p.b_hs
                               + (long long)blockIdx.x * 128 * p.ldb;

  const int tid  = threadIdx.x;
  const int lane = tid & 31;
  const int wid  = tid >> 5;
  const int wm = wid & 3;            // 4 waves along M, 32 rows each
  const int wn = wid >> 2;           // 2 waves along N, 64 cols each
  const int lr = lane & 15;
  const int hi = lane >> 4;

  // Cooperative async stage: 1024 16B transfers per 128x64 tile, 4 per thread.
  const int seg = (tid & 7) * 8;     // element offset within row (8 bf16 = 16B)
  const int r0  = tid >> 3;          // rows r0, r0+32, r0+64, r0+96
  auto issue = [&](int buf, int k0) {
#pragma unroll
    for (int q = 0; q < 4; ++q) {
      const int row = r0 + q * 32;
      unsigned lva = (unsigned)(size_t)&At[buf][row][seg];
      unsigned long long gva =
          (unsigned long long)(size_t)(A + (long long)row * p.lda + k0 + seg);
      asm volatile("global_load_async_to_lds_b128 %0, %1, off"
                   :: "v"(lva), "v"(gva) : "memory");
    }
#pragma unroll
    for (int q = 0; q < 4; ++q) {
      const int row = r0 + q * 32;
      unsigned lvb = (unsigned)(size_t)&Bt[buf][row][seg];
      unsigned long long gvb =
          (unsigned long long)(size_t)(Bm + (long long)row * p.ldb + k0 + seg);
      asm volatile("global_load_async_to_lds_b128 %0, %1, off"
                   :: "v"(lvb), "v"(gvb) : "memory");
    }
  };

  v8f acc[2][4];
#pragma unroll
  for (int i = 0; i < 2; ++i)
#pragma unroll
    for (int j = 0; j < 4; ++j)
#pragma unroll
      for (int e = 0; e < 8; ++e) acc[i][j][e] = 0.0f;

  issue(0, 0);
  const int KT = p.K >> 6;           // K-steps of 64 (all K are multiples of 64)
  for (int kt = 0; kt < KT; ++kt) {
    const int cur = kt & 1;
    if (kt + 1 < KT) {
      issue(cur ^ 1, (kt + 1) << 6);
      asm volatile("s_wait_asynccnt 0x8" ::: "memory");  // stage kt landed
    } else {
      asm volatile("s_wait_asynccnt 0x0" ::: "memory");
    }
    __syncthreads();                 // all waves' tile `cur` visible

#pragma unroll
    for (int s = 0; s < 2; ++s) {    // two 16x16x32 sub-steps
      const int ks = s * 32;
      v16bf afr[2], bfr[4];
#pragma unroll
      for (int i = 0; i < 2; ++i) {
        const bhalf* ap = &At[cur][wm * 32 + i * 16 + lr][ks + hi * 8];
        v8bf lo = *reinterpret_cast<const v8bf*>(ap);
        v8bf hs = *reinterpret_cast<const v8bf*>(ap + 16);
#pragma unroll
        for (int e = 0; e < 8; ++e) { afr[i][e] = lo[e]; afr[i][8 + e] = hs[e]; }
      }
#pragma unroll
      for (int j = 0; j < 4; ++j) {
        const bhalf* bp = &Bt[cur][wn * 64 + j * 16 + lr][ks + hi * 8];
        v8bf b0 = *reinterpret_cast<const v8bf*>(bp);
        v8bf b1 = *reinterpret_cast<const v8bf*>(bp + 16);
#pragma unroll
        for (int e = 0; e < 8; ++e) { bfr[j][e] = b0[e]; bfr[j][8 + e] = b1[e]; }
      }
#pragma unroll
      for (int i = 0; i < 2; ++i)
#pragma unroll
        for (int j = 0; j < 4; ++j)
          acc[i][j] = __builtin_amdgcn_wmma_f32_16x16x32_bf16(
              false, afr[i], false, bfr[j], (short)0, acc[i][j], false, false);
    }
    __syncthreads();                 // safe to overwrite buffer `cur` next iter
  }

  const long long cbase = (long long)bb * p.c_bs + (long long)hh * p.c_hs;
  const int m0 = blockIdx.y * 128 + wm * 32;
  const int n0 = blockIdx.x * 128 + wn * 64;
#pragma unroll
  for (int i = 0; i < 2; ++i) {
    const int rbase = m0 + i * 16 + hi * 8;
#pragma unroll
    for (int j = 0; j < 4; ++j) {
      const int col = n0 + j * 16 + lr;
      const float bv = p.bias ? p.bias[col] : 0.0f;
#pragma unroll
      for (int e = 0; e < 8; ++e) {
        const float v = acc[i][j][e] * p.alpha + bv;
        const long long off = cbase + (long long)(rbase + e) * p.ldc + col;
        if (p.Cf) p.Cf[off] = v; else p.Cb[off] = (bhalf)v;
      }
    }
  }
}

// ---------------------------------------------------------------------------
// Row softmax. One 256-thread block per row.
// s[j] = dots[r][j] + (qpe ? qpe[r][j - i_global + tm1] : 0)   for j <= jmax
// masked (causal) entries -> 0.  Output bf16 attention weights.
// ---------------------------------------------------------------------------
__global__ __launch_bounds__(256) void softmax_rows_kernel(
    const float* __restrict__ dots, const float* __restrict__ qpe,
    bhalf* __restrict__ attn, int N, int Tc, int c0, int causal, int tmem, int tm1) {
  const int r  = blockIdx.x;
  const int ig = c0 + (r % Tc);
  const int jmax = causal ? (ig + tmem) : (N - 1);
  const float* drow = dots + (long long)r * N;
  const float* prow = qpe ? qpe + (long long)r * N : nullptr;
  bhalf* arow = attn + (long long)r * N;
  const int tid = threadIdx.x;

  float cache[9];
  float mx = -3.402823466e38f;
  int idx = 0;
  for (int j = tid; j < N; j += 256, ++idx) {
    float s = -3.402823466e38f;
    if (j <= jmax) {
      s = drow[j];
      if (prow) s += prow[j - ig + tm1];
    }
    cache[idx] = s;
    mx = fmaxf(mx, s);
  }
  __shared__ float red[256];
  red[tid] = mx; __syncthreads();
  for (int o = 128; o > 0; o >>= 1) { if (tid < o) red[tid] = fmaxf(red[tid], red[tid + o]); __syncthreads(); }
  const float rowmax = red[0];
  __syncthreads();

  float sum = 0.f;
  idx = 0;
  for (int j = tid; j < N; j += 256, ++idx) {
    float e = (j <= jmax) ? __expf(cache[idx] - rowmax) : 0.f;
    cache[idx] = e;
    sum += e;
  }
  red[tid] = sum; __syncthreads();
  for (int o = 128; o > 0; o >>= 1) { if (tid < o) red[tid] += red[tid + o]; __syncthreads(); }
  const float inv = 1.0f / red[0];
  idx = 0;
  for (int j = tid; j < N; j += 256, ++idx)
    arow[j] = (bhalf)(cache[idx] * inv);
}

// ---------------------------------------------------------------------------
// LDS-tiled bf16 transpose: out[c][r] = in[r][c], batched per (b,h).
// ---------------------------------------------------------------------------
struct TransP {
  const bhalf* in; bhalf* out;
  long long i_bs, i_hs, o_bs, o_hs;
  int irs, ors, H;
};
__global__ __launch_bounds__(256) void transpose_bf16_kernel(TransP p) {
  __shared__ bhalf t[32][33];
  const int z = blockIdx.z, bb = z / p.H, hh = z % p.H;
  const bhalf* in = p.in + (long long)bb * p.i_bs + (long long)hh * p.i_hs;
  bhalf* out = p.out + (long long)bb * p.o_bs + (long long)hh * p.o_hs;
  const int c0 = blockIdx.x * 32, r0 = blockIdx.y * 32;
  const int tx = threadIdx.x, ty = threadIdx.y;
#pragma unroll
  for (int s = 0; s < 32; s += 8) t[ty + s][tx] = in[(long long)(r0 + ty + s) * p.irs + c0 + tx];
  __syncthreads();
#pragma unroll
  for (int s = 0; s < 32; s += 8) out[(long long)(c0 + ty + s) * p.ors + r0 + tx] = t[tx][ty + s];
}

// ---------------------------------------------------------------------------
// Deterministic two-stage squared-difference mean reduction.
// ---------------------------------------------------------------------------
__global__ __launch_bounds__(256) void sqdiff_kernel(const float* __restrict__ a,
                                                     const float* __restrict__ b,
                                                     float* __restrict__ part, long long n) {
  float s = 0.f;
  for (long long i = (long long)blockIdx.x * 256 + threadIdx.x; i < n; i += (long long)gridDim.x * 256) {
    float d = a[i] - b[i]; s += d * d;
  }
  __shared__ float red[256];
  red[threadIdx.x] = s; __syncthreads();
  for (int o = 128; o > 0; o >>= 1) { if (threadIdx.x < o) red[threadIdx.x] += red[threadIdx.x + o]; __syncthreads(); }
  if (threadIdx.x == 0) part[blockIdx.x] = red[0];
}
__global__ __launch_bounds__(256) void aux_final_kernel(const float* __restrict__ part,
                                                        float* __restrict__ outp, int np, float scale) {
  float s = 0.f;
  for (int i = threadIdx.x; i < np; i += 256) s += part[i];
  __shared__ float red[256];
  red[threadIdx.x] = s; __syncthreads();
  for (int o = 128; o > 0; o >>= 1) { if (threadIdx.x < o) red[threadIdx.x] += red[threadIdx.x + o]; __syncthreads(); }
  if (threadIdx.x == 0) outp[0] = red[0] * scale;
}

// ---------------------------------------------------------------------------
// Host side
// ---------------------------------------------------------------------------
static inline void launch_gemm(hipStream_t s,
    const bhalf* A, long long a_bs, long long a_hs, int lda,
    const bhalf* B, long long b_bs, long long b_hs, int ldb,
    float* Cf, bhalf* Cb, long long c_bs, long long c_hs, int ldc,
    const float* bias, int M, int N, int K, int Z, int H, float alpha) {
  GemmP p{A, B, Cf, Cb, bias, a_bs, a_hs, b_bs, b_hs, c_bs, c_hs, lda, ldb, ldc, K, H, alpha};
  dim3 g((unsigned)(N / 128), (unsigned)(M / 128), (unsigned)Z);
  wmma_gemm_kernel<<<g, 256, 0, s>>>(p);
}

extern "C" void kernel_launch(void* const* d_in, const int* in_sizes, int n_in,
                              void* d_out, int out_size, void* d_ws, size_t ws_size,
                              hipStream_t stream) {
  (void)in_sizes; (void)n_in; (void)out_size; (void)ws_size;
  const float* x     = (const float*)d_in[0];
  const float* mem   = (const float*)d_in[1];
  const float* cmem  = (const float*)d_in[2];
  const float* pos   = (const float*)d_in[3];
  const float* w_q   = (const float*)d_in[4];
  const float* w_kv  = (const float*)d_in[5];
  const float* w_out = (const float*)d_in[6];
  const float* b_out = (const float*)d_in[7];
  const float* convw = (const float*)d_in[8];
  const float* convb = (const float*)d_in[9];

  float* logits   = (float*)d_out;            // 4*1024*1024
  float* new_mem  = logits + 4194304;         // 4*1024*1024
  float* new_cmem = new_mem + 4194304;        // 4*256*1024
  float* aux      = new_cmem + 1048576;       // scalar

  char* ws = (char*)d_ws;
  size_t off = 0;
  auto take = [&](size_t bytes) -> char* {
    char* p = ws + off;
    off = (off + bytes + 255) & ~(size_t)255;
    return p;
  };
  bhalf* wq_bf   = (bhalf*)take((size_t)1048576 * 2);
  bhalf* wkv_bf  = (bhalf*)take((size_t)2097152 * 2);
  bhalf* wout_bf = (bhalf*)take((size_t)1048576 * 2);
  bhalf* cw_bf   = (bhalf*)take((size_t)4194304 * 2);
  bhalf* x_bf    = (bhalf*)take((size_t)4194304 * 2);
  bhalf* kvin_bf = (bhalf*)take((size_t)9437184 * 2);   // (B, 2304, E) = [cmem|mem|x]
  bhalf* pe_bf   = (bhalf*)take((size_t)2359296 * 2);   // (H, 2304, 128)
  bhalf* q_bf    = (bhalf*)take((size_t)4194304 * 2);   // (B, T, E)
  bhalf* kv_bf   = (bhalf*)take((size_t)18874368 * 2);  // (B, 2304, 2048)  [k|v]
  bhalf* vT_bf   = (bhalf*)take((size_t)9437184 * 2);   // (B, H, 128, 2304)
  bhalf* comp_bf = (bhalf*)take((size_t)1048576 * 2);   // (B, 256, E)
  bhalf* ckv_bf  = (bhalf*)take((size_t)2097152 * 2);   // (B, 256, 2048)
  bhalf* cvT_bf  = (bhalf*)take((size_t)1048576 * 2);   // (B, H, 128, 256)
  bhalf* out_bf  = (bhalf*)take((size_t)4194304 * 2);   // (B, T, E) merged attention out
  float* dots    = (float*)take((size_t)9437184 * 4);   // chunk scores (B,H,128,<=2304)
  float* qpe     = (float*)take((size_t)9437184 * 4);   // chunk q.pe^T
  bhalf* attn_bf = (bhalf*)take((size_t)9437184 * 2);   // chunk attn weights
  float* out1    = (float*)take((size_t)4194304 * 4);   // aux attn(mem)
  float* out2    = (float*)take((size_t)4194304 * 4);   // aux attn(cmem)
  float* part    = (float*)take((size_t)1024 * 4);

  const float SCALEf = 0.08838834764831845f;  // 128^-0.5

  auto cvt = [&](const float* s, bhalf* d, long long n, int nb, long long sbs, long long dbs) {
    dim3 g((unsigned)((n + 255) / 256), (unsigned)nb);
    cvt_f32_bf16_kernel<<<g, 256, 0, stream>>>(s, d, n, sbs, dbs);
  };

  // ---- bf16 staging -------------------------------------------------------
  cvt(w_q,  wq_bf,   1048576, 1, 0, 0);
  cvt(w_kv, wkv_bf,  2097152, 1, 0, 0);
  cvt(w_out, wout_bf, 1048576, 1, 0, 0);
  cvt(pos,  pe_bf,   2359296, 1, 0, 0);
  cvt(x,    x_bf,    4194304, 1, 0, 0);
  cvt(cmem, kvin_bf,            262144,  4, 262144,  2359296);
  cvt(mem,  kvin_bf + 262144,   1048576, 4, 1048576, 2359296);
  cvt(x,    kvin_bf + 1310720,  1048576, 4, 1048576, 2359296);
  repack_convw_kernel<<<(4194304 + 255) / 256, 256, 0, stream>>>(convw, cw_bf, 4194304);

  // ---- projections --------------------------------------------------------
  // q = x @ w_q^T            (4096,1024,1024)
  launch_gemm(stream, x_bf, 0, 0, 1024, wq_bf, 0, 0, 1024,
              nullptr, q_bf, 0, 0, 1024, nullptr, 4096, 1024, 1024, 1, 1, 1.f);
  // kv = [cmem|mem|x] @ w_kv^T  (9216,2048,1024)
  launch_gemm(stream, kvin_bf, 0, 0, 1024, wkv_bf, 0, 0, 1024,
              nullptr, kv_bf, 0, 0, 2048, nullptr, 9216, 2048, 1024, 1, 1, 1.f);
  // vT[b,h] = v[b,h]^T  (2304x128 -> 128x2304)
  {
    TransP tp{kv_bf + 1024, vT_bf, 4718592, 128, 2359296, 294912, 2048, 2304, 8};
    transpose_bf16_kernel<<<dim3(4, 72, 32), dim3(32, 8), 0, stream>>>(tp);
  }
  // conv compress = GEMM: mem[b] as [256,4096] x conv_w_rep^T + conv_b -> new_cmem (f32)
  launch_gemm(stream, kvin_bf + 262144, 2359296, 0, 4096, cw_bf, 0, 0, 4096,
              new_cmem, nullptr, 262144, 0, 1024, convb, 256, 1024, 4096, 4, 1, 1.f);
  cvt(new_cmem, comp_bf, 1048576, 1, 0, 0);
  // ckv = compressed @ w_kv^T  (1024,2048,1024)
  launch_gemm(stream, comp_bf, 0, 0, 1024, wkv_bf, 0, 0, 1024,
              nullptr, ckv_bf, 0, 0, 2048, nullptr, 1024, 2048, 1024, 1, 1, 1.f);
  {
    TransP tp{ckv_bf + 1024, cvT_bf, 524288, 128, 262144, 32768, 2048, 256, 8};
    transpose_bf16_kernel<<<dim3(4, 8, 32), dim3(32, 8), 0, stream>>>(tp);
  }

  // ---- main attention, chunked over T (Tc=128, 8 chunks) ------------------
  for (int c = 0; c < 8; ++c) {
    const bhalf* qc = q_bf + (long long)c * 131072;  // c*Tc*E
    // dots = SCALE * q k^T   (per b,h: 128 x 2304, K=128)
    launch_gemm(stream, qc, 1048576, 128, 1024, kv_bf, 4718592, 128, 2048,
                dots, nullptr, 2359296, 294912, 2304, nullptr, 128, 2304, 128, 32, 8, SCALEf);
    // qpe = SCALE * q pe^T   (pe shared across batch)
    launch_gemm(stream, qc, 1048576, 128, 1024, pe_bf, 0, 294912, 128,
                qpe, nullptr, 2359296, 294912, 2304, nullptr, 128, 2304, 128, 32, 8, SCALEf);
    // fused shift + causal mask + softmax -> bf16 attn
    softmax_rows_kernel<<<4096, 256, 0, stream>>>(dots, qpe, attn_bf, 2304, 128, c * 128, 1, 1280, 1023);
    // out = attn @ v         (128 x 128, K=2304) -> merged (B,T,E) bf16
    launch_gemm(stream, attn_bf, 2359296, 294912, 2304, vT_bf, 2359296, 294912, 2304,
                nullptr, out_bf + (long long)c * 131072, 1048576, 128, 1024,
                nullptr, 128, 128, 2304, 32, 8, 1.f);
  }
  // logits = out @ w_out^T + b_out
  launch_gemm(stream, out_bf, 0, 0, 1024, wout_bf, 0, 0, 1024,
              logits, nullptr, 0, 0, 1024, b_out, 4096, 1024, 1024, 1, 1, 1.f);
  // new_mem = x
  hipMemcpyAsync(new_mem, x, (size_t)4194304 * 4, hipMemcpyDeviceToDevice, stream);

  // ---- aux loss: full attn vs mem k/v and cmem k/v ------------------------
  for (int c = 0; c < 8; ++c) {
    const bhalf* qc = q_bf + (long long)c * 131072;
    // attn over old_k (mem rows 256..1279 of kv)
    launch_gemm(stream, qc, 1048576, 128, 1024, kv_bf + 524288, 4718592, 128, 2048,
                dots, nullptr, 1048576, 131072, 1024, nullptr, 128, 1024, 128, 32, 8, SCALEf);
    softmax_rows_kernel<<<4096, 256, 0, stream>>>(dots, nullptr, attn_bf, 1024, 128, 0, 0, 0, 0);
    launch_gemm(stream, attn_bf, 1048576, 131072, 1024, vT_bf + 256, 2359296, 294912, 2304,
                out1 + (long long)c * 16384, nullptr, 1048576, 131072, 128,
                nullptr, 128, 128, 1024, 32, 8, 1.f);
    // attn over cmem_k
    launch_gemm(stream, qc, 1048576, 128, 1024, ckv_bf, 524288, 128, 2048,
                dots, nullptr, 262144, 32768, 256, nullptr, 128, 256, 128, 32, 8, SCALEf);
    softmax_rows_kernel<<<4096, 256, 0, stream>>>(dots, nullptr, attn_bf, 256, 128, 0, 0, 0, 0);
    launch_gemm(stream, attn_bf, 262144, 32768, 256, cvT_bf, 262144, 32768, 256,
                out2 + (long long)c * 16384, nullptr, 1048576, 131072, 128,
                nullptr, 128, 128, 256, 32, 8, 1.f);
  }
  sqdiff_kernel<<<1024, 256, 0, stream>>>(out1, out2, part, 4194304);
  aux_final_kernel<<<1, 256, 0, stream>>>(part, aux, 1024, 1.0f / 4194304.0f);
}
